// CRVAE_12601434047102
// MI455X (gfx1250) — compile-verified, matrix operand order checked by reading, not run on hardware
//
#include <hip/hip_runtime.h>

// ---------------------------------------------------------------------------
// CR-VAE on MI455X (gfx1250, wave32, WMMA).
// Latency-bound sequential GRU scans -> one workgroup per decoder series,
// recurrent state in LDS, fused [x|h|1] x [Wih|Whh|b]^T WMMA (bf16 in /
// f32 accum, K=320 = 10 k-steps of v_wmma_f32_16x16x32_bf16).
// The K-chain is a *runtime* loop (first/last k-step peeled) so the post-RA
// scheduler cannot hoist the whole chain's B-fragment loads at once -- a
// fully unrolled chain made it queue 38 global loads (320 VGPRs of B data)
// and spill to scratch. 16 accumulators (128 VGPRs) + 3-4 live B-frags fit
// the 256-VGPR budget with no spills.
// ---------------------------------------------------------------------------

typedef unsigned short u16;
typedef unsigned int   u32;
typedef __attribute__((ext_vector_type(16))) __bf16 v16bf;
typedef __attribute__((ext_vector_type(8)))  float  v8f;
typedef __attribute__((ext_vector_type(4)))  u32    u32x4;

union Frag { v16bf v; u32x4 q[2]; u16 s[16]; };

#define Bb   64
#define Tt   64
#define Pp   32
#define Hh   256
#define AROW 320            // A row stride: 32(x)+256(h)+1(bias)+31(pad) bf16
#define NSER_PW 33          // 32 decoder series + 1 encoder slot
#define TDEC 54
#define GSTRIDE 81920       // elems per gate-pack: 16 jt * 10 ks * 512
#define SMEM_BYTES 84992    // 2*64*320*2 (bufA/bufB) + 1024 (linw) + 2048 (partial)

__device__ __forceinline__ u16 f2bf(float f) {
    u32 u = __float_as_uint(f);
    u32 r = (u + 0x7FFFu + ((u >> 16) & 1u)) >> 16;   // round-to-nearest-even
    return (u16)r;
}
__device__ __forceinline__ float bf2f(u16 h) { return __uint_as_float(((u32)h) << 16); }
__device__ __forceinline__ float sigm(float x)  { return 1.f / (1.f + __expf(-x)); }
__device__ __forceinline__ float tanh_(float x) { float e = __expf(-2.f * x); return (1.f - e) / (1.f + e); }

__device__ __forceinline__ v16bf ld_b_frag(const u16* p) {      // 16 contiguous bf16
    Frag f; f.q[0] = *(const u32x4*)(p); f.q[1] = *(const u32x4*)(p + 8); return f.v;
}
__device__ __forceinline__ v16bf ld_a_frag(const u16* p) {      // K 0..7 and K 16..23 chunks
    Frag f; f.q[0] = *(const u32x4*)(p); f.q[1] = *(const u32x4*)(p + 16); return f.v;
}
__device__ __forceinline__ v8f wmma_bf16(v16bf a, v16bf b, v8f c) {
    return __builtin_amdgcn_wmma_f32_16x16x32_bf16(false, a, false, b, (short)0, c, false, false);
}

// ---------------------------------------------------------------------------
// Pack weights into WMMA B-operand tiles (bf16): one 32x16 KxN tile = 512
// elems, lane L reads 16 contiguous bf16 at tile + L*16. Gate-packs g:
// 0=r, 1=z (full K incl. summed bias), 2=n_h (Whh_n+bhh_n), 3=n_x
// (Wih_n+bih_n). K map matches the ISA 16-bit A/B layout: lanes 0-15 elems
// 0..7 -> K 0..7, 8..15 -> K 16..23; lanes 16-31: +8.
// ---------------------------------------------------------------------------
__global__ void CRVAE_pack(const float* __restrict__ Wih_d, const float* __restrict__ Whh_d,
                           const float* __restrict__ bih_d, const float* __restrict__ bhh_d,
                           const float* __restrict__ Wih_e, const float* __restrict__ Whh_e,
                           const float* __restrict__ bih_e, const float* __restrict__ bhh_e,
                           u16* __restrict__ PW) {
    int idx = blockIdx.x * blockDim.x + threadIdx.x;
    const int total = NSER_PW * 4 * 16 * 10 * 512;
    if (idx >= total) return;
    int te = idx & 511; int lane = te >> 4; int e = te & 15;
    int rest = idx >> 9;
    int ks = rest % 10; rest /= 10;
    int jt = rest & 15; rest >>= 4;
    int g  = rest & 3;  int p = rest >> 2;

    int K  = ((e < 8) ? 0 : 16) + (lane >> 4) * 8 + (e & 7);
    int kk = ks * 32 + K;
    int row = ((g == 0) ? 0 : (g == 1) ? 256 : 512) + jt * 16 + (lane & 15);

    const bool dec = (p < Pp);
    const float* Wih = dec ? Wih_d + (size_t)p * 768 * 32  : Wih_e;
    const float* Whh = dec ? Whh_d + (size_t)p * 768 * 256 : Whh_e;
    const float* bih = dec ? bih_d + (size_t)p * 768       : bih_e;
    const float* bhh = dec ? bhh_d + (size_t)p * 768       : bhh_e;

    float val = 0.f;
    if (g <= 1) {                                   // r, z: fully fused
        if (kk < 32)        val = Wih[row * 32 + kk];
        else if (kk < 288)  val = Whh[row * 256 + (kk - 32)];
        else if (kk == 288) val = bih[row] + bhh[row];
    } else if (g == 2) {                            // n: h-side
        if (kk >= 32 && kk < 288) val = Whh[row * 256 + (kk - 32)];
        else if (kk == 288)       val = bhh[row];
    } else {                                        // n: x-side
        if (kk < 32)        val = Wih[row * 32 + kk];
        else if (kk == 288) val = bih[row];
    }
    PW[idx] = f2bf(val);
}

// ---------------------------------------------------------------------------
// Fused GRU scan. One workgroup = one series (or the encoder). 512 threads =
// 16 wave32; wave w owns gate-column tile jt = w for all M-tiles, so
// r/z/gh_n/gi_n for any (b,j) sit in the same lane/VGPR slot and the GRU
// epilogue is pure VALU on accumulators. h double-buffers through LDS in the
// WMMA A layout; x_{t+1} is staged concurrently with the step-t GEMM.
// ---------------------------------------------------------------------------
__global__ void __launch_bounds__(512)
CRVAE_gru_scan(const float* __restrict__ X,
               const float* __restrict__ h0,        // fp32 [64*256] or null
               const u16*   __restrict__ PW,
               int pw_base, int nsteps, int x_mode,  // 0=encoder, 1=decoder
               const float* __restrict__ lin_w, const float* __restrict__ lin_b,
               float* __restrict__ pred_out, float* __restrict__ hout) {
    extern __shared__ char smem[];
    u16*   bufA    = (u16*)smem;                    // [64][320] bf16
    u16*   bufB    = bufA + Bb * AROW;
    float* linw    = (float*)(smem + 81920);        // [256]
    float* partial = (float*)(smem + 82944);        // [64*8]

    const int tid  = threadIdx.x;
    const int lane = tid & 31;
    const int w    = tid >> 5;                      // wave id == jt, 0..15
    const int hi   = lane >> 4;
    const int ml   = lane & 15;
    const int p    = pw_base + blockIdx.x;

    // ---- init: bufA = [x_0 | h_0 | 1 | 0], bufB = [* | * | 1 | 0] ----
    for (int i = tid; i < Bb * AROW; i += blockDim.x) {
        int row = i / AROW, col = i % AROW;
        u16 va = 0;
        if (col < 32) {
            float xv = (x_mode == 0) ? X[(row * Tt + 0) * Pp + col] : 0.f;
            va = f2bf(xv);
        } else if (col < 288) {
            va = f2bf(h0 ? h0[row * Hh + (col - 32)] : 0.f);
        } else if (col == 288) {
            va = (u16)0x3F80;                        // 1.0 bf16 (bias column)
        }
        bufA[i] = va;
        bufB[i] = (col == 288) ? (u16)0x3F80 : (u16)0;
    }
    if (lin_w && tid < Hh) linw[tid] = lin_w[p * Hh + tid];
    __syncthreads();

    u16* cur = bufA;
    u16* nxt = bufB;
    const u16* pwp = PW + (size_t)p * 4 * GSTRIDE + (size_t)(w * 10) * 512;

    for (int t = 0; t < nsteps; ++t) {
        // stage x_{t+1} into nxt (disjoint from epilogue's h region)
        int tnext = t + 1;
        for (int i = tid; i < Bb * Pp; i += blockDim.x) {
            int b = i >> 5, c = i & 31;
            int xt = (x_mode == 0) ? tnext : (9 + tnext);       // dec_in[:,t] = X[:,9+t], t>=1
            nxt[b * AROW + c] = f2bf(X[(b * Tt + xt) * Pp + c]);
        }

        // ---- fused GEMM: 16 accumulators, runtime K loop (peeled ends) ----
        v8f ar[4], az[4], anh[4], anx[4];

        {   // ks = 0: x-columns -> r, z, n_x tiles (n_h starts at zero)
            const u16* bt = pwp + lane * 16;
            v16bf br = ld_b_frag(bt);
            v16bf bz = ld_b_frag(bt + 1 * GSTRIDE);
            v16bf bx = ld_b_frag(bt + 3 * GSTRIDE);
            #pragma unroll
            for (int mt = 0; mt < 4; ++mt) {
                v8f zacc;
                #pragma unroll
                for (int i = 0; i < 8; ++i) { zacc[i] = 0.f; anh[mt][i] = 0.f; }
                v16bf a = ld_a_frag(cur + (size_t)(mt * 16 + ml) * AROW + hi * 8);
                ar[mt]  = wmma_bf16(a, br, zacc);
                az[mt]  = wmma_bf16(a, bz, zacc);
                anx[mt] = wmma_bf16(a, bx, zacc);
            }
        }

        #pragma unroll 1
        for (int ks = 1; ks < 9; ++ks) {            // h-columns: r, z, n_h
            const u16* bt = pwp + ks * 512 + lane * 16;
            v16bf br = ld_b_frag(bt);
            v16bf bz = ld_b_frag(bt + 1 * GSTRIDE);
            v16bf bn = ld_b_frag(bt + 2 * GSTRIDE);
            #pragma unroll
            for (int mt = 0; mt < 4; ++mt) {
                v16bf a = ld_a_frag(cur + (size_t)(mt * 16 + ml) * AROW + ks * 32 + hi * 8);
                ar[mt]  = wmma_bf16(a, br, ar[mt]);
                az[mt]  = wmma_bf16(a, bz, az[mt]);
                anh[mt] = wmma_bf16(a, bn, anh[mt]);
            }
        }

        {   // ks = 9: bias column -> all four gate packs
            const u16* bt = pwp + 9 * 512 + lane * 16;
            v16bf br = ld_b_frag(bt);
            v16bf bz = ld_b_frag(bt + 1 * GSTRIDE);
            v16bf bn = ld_b_frag(bt + 2 * GSTRIDE);
            v16bf bx = ld_b_frag(bt + 3 * GSTRIDE);
            #pragma unroll
            for (int mt = 0; mt < 4; ++mt) {
                v16bf a = ld_a_frag(cur + (size_t)(mt * 16 + ml) * AROW + 9 * 32 + hi * 8);
                ar[mt]  = wmma_bf16(a, br, ar[mt]);
                az[mt]  = wmma_bf16(a, bz, az[mt]);
                anh[mt] = wmma_bf16(a, bn, anh[mt]);
                anx[mt] = wmma_bf16(a, bx, anx[mt]);
            }
        }

        // ---- GRU epilogue: C/D layout -> lanes 0-15: VGPR r = M r; 16-31: M 8+r
        #pragma unroll
        for (int mt = 0; mt < 4; ++mt) {
            #pragma unroll
            for (int r = 0; r < 8; ++r) {
                int row = mt * 16 + r + 8 * hi;
                int col = 32 + w * 16 + ml;
                float ho = bf2f(cur[row * AROW + col]);
                float rg = sigm(ar[mt][r]);
                float zg = sigm(az[mt][r]);
                float nn = tanh_(anx[mt][r] + rg * anh[mt][r]);
                float hn = (1.f - zg) * nn + zg * ho;
                nxt[row * AROW + col] = f2bf(hn);
            }
        }
        __syncthreads();                                // h_{t+1} complete in nxt

        if (pred_out) {                                 // linear head on hs[:,t] = h_{t+1}
            int b = tid & 63, part = tid >> 6;          // 8 partial chunks of 32
            float s = 0.f;
            #pragma unroll 8
            for (int jj = part * 32; jj < part * 32 + 32; ++jj)
                s += bf2f(nxt[b * AROW + 32 + jj]) * linw[jj];
            partial[b * 8 + part] = s;
            __syncthreads();
            if (tid < Bb) {
                float s2 = lin_b[p];
                #pragma unroll
                for (int q = 0; q < 8; ++q) s2 += partial[tid * 8 + q];
                pred_out[(size_t)p * (Bb * TDEC) + tid * TDEC + t] = s2;
            }
        }
        u16* tmp = cur; cur = nxt; nxt = tmp;
    }

    if (hout) {
        __syncthreads();
        for (int i = tid; i < Bb * Hh; i += blockDim.x) {
            int b = i >> 8, j = i & 255;
            hout[i] = bf2f(cur[b * AROW + 32 + j]);
        }
    }
}

// ---------------------------------------------------------------------------
// Latent head: mu/log_var (256x256 GEMVs, L2-resident) + reparameterization.
// ---------------------------------------------------------------------------
__global__ void CRVAE_latent(const float* __restrict__ h,
                             const float* __restrict__ muw, const float* __restrict__ mub,
                             const float* __restrict__ sw,  const float* __restrict__ sb,
                             const float* __restrict__ eps,
                             float* __restrict__ mu_out, float* __restrict__ lv_out,
                             float* __restrict__ z_out) {
    int idx = blockIdx.x * blockDim.x + threadIdx.x;
    if (idx >= Bb * Hh) return;
    int b = idx >> 8, j = idx & 255;
    const float* hv = h + b * Hh;
    float m = mub[j], l = sb[j];
    for (int k = 0; k < Hh; ++k) {
        float hk = hv[k];
        m += muw[j * Hh + k] * hk;
        l += sw[j * Hh + k] * hk;
    }
    mu_out[idx] = m;
    lv_out[idx] = l;
    z_out[idx]  = m + __expf(0.5f * l) * eps[idx];
}

extern "C" void kernel_launch(void* const* d_in, const int* in_sizes, int n_in,
                              void* d_out, int out_size, void* d_ws, size_t ws_size,
                              hipStream_t stream) {
    const float* X     = (const float*)d_in[0];
    const float* eps   = (const float*)d_in[1];
    const float* Wih_e = (const float*)d_in[2];
    const float* Whh_e = (const float*)d_in[3];
    const float* bih_e = (const float*)d_in[4];
    const float* bhh_e = (const float*)d_in[5];
    const float* muw   = (const float*)d_in[6];
    const float* mub   = (const float*)d_in[7];
    const float* sw    = (const float*)d_in[8];
    const float* sb    = (const float*)d_in[9];
    const float* Wih_d = (const float*)d_in[10];
    const float* Whh_d = (const float*)d_in[11];
    const float* bih_d = (const float*)d_in[12];
    const float* bhh_d = (const float*)d_in[13];
    const float* lin_w = (const float*)d_in[14];
    const float* lin_b = (const float*)d_in[15];

    // workspace: packed bf16 weights (~20.6 MB, L2-resident) + h_t + z
    const size_t PW_ELEMS = (size_t)NSER_PW * 4 * 16 * 10 * 512;   // 10,813,440
    u16*   PW   = (u16*)d_ws;
    float* h_t  = (float*)((char*)d_ws + PW_ELEMS * sizeof(u16));
    float* zbuf = h_t + Bb * Hh;

    float* pred = (float*)d_out;                 // [32,64,54,1]
    float* lv   = pred + Pp * Bb * TDEC;         // [1,64,256]
    float* mu   = lv + Bb * Hh;                  // [1,64,256]

    const int pack_total = (int)PW_ELEMS;
    CRVAE_pack<<<(pack_total + 255) / 256, 256, 0, stream>>>(
        Wih_d, Whh_d, bih_d, bhh_d, Wih_e, Whh_e, bih_e, bhh_e, PW);

    // encoder: 1 WG, 10 steps, h0 = 0, PW slot 32
    CRVAE_gru_scan<<<1, 512, SMEM_BYTES, stream>>>(
        X, nullptr, PW, /*pw_base=*/Pp, /*nsteps=*/10, /*x_mode=*/0,
        nullptr, nullptr, nullptr, h_t);

    CRVAE_latent<<<(Bb * Hh + 255) / 256, 256, 0, stream>>>(
        h_t, muw, mub, sw, sb, eps, mu, lv, zbuf);

    // decoders: 32 WGs (one per series), 54 steps, h0 = z
    CRVAE_gru_scan<<<Pp, 512, SMEM_BYTES, stream>>>(
        X, zbuf, PW, /*pw_base=*/0, /*nsteps=*/TDEC, /*x_mode=*/1,
        lin_w, lin_b, pred, nullptr);
}
